// Relation_18803366822301
// MI455X (gfx1250) — matrix operand order repaired and checked by reading, not compiled, not run on hardware
//
#include <hip/hip_runtime.h>

// ---------------------------------------------------------------------------
// Relation block, restructured:
//   att_i = ( alpha_i @ (beta^T @ unary) - (alpha_i . beta_i) * unary_i ) / N
//   out   = att @ Wr^T + x
// O(B*N*F^2) instead of O(B*N^2*F); no 64MB score tensor.
// All GEMMs run on v_wmma_f32_16x16x32_bf16 (wave32), f32 accumulate.
// ---------------------------------------------------------------------------

#define BB 16
#define NN 1024
#define FF 256
#define NT (NN / 16) // 64 token tiles
#define GT (FF / 16) // 16 feature tiles

typedef __bf16 bf16_t;
typedef __attribute__((ext_vector_type(16))) __bf16 v16bf;
typedef __attribute__((ext_vector_type(8)))  float  v8f;
typedef __attribute__((ext_vector_type(4)))  unsigned int u32x4; // trivial, union-safe

union Frag {
    v16bf v;
    u32x4 q[2];
};

__device__ __forceinline__ v8f wmma_bf16(const Frag& a, const Frag& b, v8f c) {
    // (neg_a, A, neg_b, B, c_mod, C, reuse_a, reuse_b)
    return __builtin_amdgcn_wmma_f32_16x16x32_bf16(false, a.v, false, b.v,
                                                   (short)0, c, false, false);
}

// A-fragment (16x32, row-major source). ISA 7.12.2 16-bit A layout:
// lane = 16h+m holds row m: VGPR0-3 = K[8h..8h+7], VGPR4-7 = K[16+8h..16+8h+7]
// -> two contiguous 16-byte loads per lane.
__device__ __forceinline__ Frag load_a_frag(const bf16_t* row_base, int k, int h) {
    Frag f;
    f.q[0] = *(const u32x4*)(row_base + k + 8 * h);
    f.q[1] = *(const u32x4*)(row_base + k + 16 + 8 * h);
    return f;
}

// B-fragment (32x16) sourced from the N-major (transposed) matrix:
// lane = 16h+n holds column n: K[8h..8h+7], K[16+8h..16+8h+7] -> two 16-byte loads
// from row n of B^T.
__device__ __forceinline__ Frag load_b_frag(const bf16_t* col_base, int k, int h) {
    Frag f;
    f.q[0] = *(const u32x4*)(col_base + k + 8 * h);
    f.q[1] = *(const u32x4*)(col_base + k + 16 + 8 * h);
    return f;
}

// ---------------------------------------------------------------------------
// K0: f32 -> bf16 conversion (x and the four weight matrices)
// ---------------------------------------------------------------------------
__global__ void cvt_bf16_kernel(const float* __restrict__ src,
                                bf16_t* __restrict__ dst, int n) {
    int i = blockIdx.x * blockDim.x + threadIdx.x;
    if (i < n) dst[i] = (bf16_t)src[i];
}

// ---------------------------------------------------------------------------
// K1: projections. grid = (B*NT, 3), block = 256 (8 waves), wave -> 2 gtiles.
// alpha -> [B,N,F] bf16
// beta  -> [B,N,F] bf16 and betaT [B,F,N] bf16
// unary -> [B,N,F] f32 (for exact diagonal correction) and unaryT [B,F,N] bf16
// ---------------------------------------------------------------------------
__global__ __launch_bounds__(256) void proj_kernel(
    const bf16_t* __restrict__ xb,
    const bf16_t* __restrict__ Wa, const bf16_t* __restrict__ Wb,
    const bf16_t* __restrict__ Wu, const float* __restrict__ bias,
    bf16_t* __restrict__ alpha, bf16_t* __restrict__ beta,
    bf16_t* __restrict__ betaT, float* __restrict__ unary_f,
    bf16_t* __restrict__ unaryT) {
    const int b = blockIdx.x / NT, nt = blockIdx.x % NT;
    const int proj = blockIdx.y;
    const int wave = threadIdx.x >> 5, lane = threadIdx.x & 31;
    const int h = lane >> 4, l16 = lane & 15;
    const bf16_t* W = (proj == 0) ? Wa : ((proj == 1) ? Wb : Wu);

    const int tokA = nt * 16 + l16;
    const bf16_t* arow = xb + (size_t)(b * NN + tokA) * FF;

    for (int gi = 0; gi < 2; ++gi) {
        const int gt = wave * 2 + gi;
        // y = x @ W^T : B-matrix = W^T, whose columns are rows of W (row-major).
        const bf16_t* brow = W + (size_t)(gt * 16 + l16) * FF;
        v8f acc = {};
        for (int k = 0; k < FF; k += 32) {
            Frag a = load_a_frag(arow, k, h);
            Frag bfr = load_b_frag(brow, k, h);
            acc = wmma_bf16(a, bfr, acc);
        }
        // D layout: VGPR r, lane 16h+n -> element [M = r+8h][N = n]
        const int g = gt * 16 + l16;
        for (int r = 0; r < 8; ++r) {
            const int tok = nt * 16 + r + 8 * h;
            const size_t idx = (size_t)(b * NN + tok) * FF + g;
            float v = acc[r];
            if (proj == 0) {
                alpha[idx] = (bf16_t)v;
            } else if (proj == 1) {
                beta[idx] = (bf16_t)v;
                betaT[(size_t)(b * FF + g) * NN + tok] = (bf16_t)v;
            } else {
                v += bias[g];
                unary_f[idx] = v;
                unaryT[(size_t)(b * FF + g) * NN + tok] = (bf16_t)v;
            }
        }
    }
}

// ---------------------------------------------------------------------------
// K2: d[b,i] = alpha_i . beta_i  (per-token diagonal score)
// ---------------------------------------------------------------------------
__global__ void diag_kernel(const bf16_t* __restrict__ alpha,
                            const bf16_t* __restrict__ beta,
                            float* __restrict__ d) {
    const int t = blockIdx.x * blockDim.x + threadIdx.x; // t in [0, B*N)
    const bf16_t* ap = alpha + (size_t)t * FF;
    const bf16_t* bp = beta + (size_t)t * FF;
    float s = 0.f;
#pragma unroll 8
    for (int f = 0; f < FF; ++f) s += (float)ap[f] * (float)bp[f];
    d[t] = s;
}

// ---------------------------------------------------------------------------
// K3: P = M^T = unary^T @ beta  (per batch, F x F, reduction over N=1024).
// A = unaryT rows (contiguous), B columns = betaT rows (contiguous).
// grid = (B, 32), block = 256 -> 8 waves x 32 blocks = 256 tiles per batch.
// ---------------------------------------------------------------------------
__global__ __launch_bounds__(256) void pmat_kernel(
    const bf16_t* __restrict__ unaryT, const bf16_t* __restrict__ betaT,
    bf16_t* __restrict__ P) {
    const int b = blockIdx.x;
    const int wave = threadIdx.x >> 5, lane = threadIdx.x & 31;
    const int h = lane >> 4, l16 = lane & 15;
    const int tile = blockIdx.y * 8 + wave;
    const int tm = tile >> 4, tn = tile & 15;

    const bf16_t* arow = unaryT + (size_t)(b * FF + tm * 16 + l16) * NN;
    const bf16_t* brow = betaT + (size_t)(b * FF + tn * 16 + l16) * NN;
    v8f acc = {};
    for (int k = 0; k < NN; k += 32) {
        Frag a = load_a_frag(arow, k, h);
        Frag bfr = load_b_frag(brow, k, h);
        acc = wmma_bf16(a, bfr, acc);
    }
    const int g1 = tn * 16 + l16;
    for (int r = 0; r < 8; ++r) {
        const int g2 = tm * 16 + r + 8 * h;
        P[(size_t)(b * FF + g2) * FF + g1] = (bf16_t)acc[r];
    }
}

// ---------------------------------------------------------------------------
// K4: fused  att = (alpha @ P^T - d*unary)/N  -> LDS ->  out = att @ Wr^T + x
// grid = B*NT, block = 256 (8 waves). att tile = 16 tokens x 256 feats in LDS.
// ---------------------------------------------------------------------------
__global__ __launch_bounds__(256) void final_kernel(
    const bf16_t* __restrict__ alpha, const bf16_t* __restrict__ P,
    const float* __restrict__ dvec, const float* __restrict__ unary_f,
    const bf16_t* __restrict__ Wr, const float* __restrict__ x,
    float* __restrict__ out) {
    __shared__ bf16_t att[16 * FF]; // 8 KB of the 320 KB WGP LDS

    const int b = blockIdx.x / NT, nt = blockIdx.x % NT;
    const int wave = threadIdx.x >> 5, lane = threadIdx.x & 31;
    const int h = lane >> 4, l16 = lane & 15;
    const int tokA = nt * 16 + l16;
    const float inv_n = 1.0f / (float)NN;

    // Phase 1: att[i][g] = (sum_f alpha[i][f]*P[g][f] - d_i*u_i[g]) / N
    const bf16_t* arow = alpha + (size_t)(b * NN + tokA) * FF;
    for (int gi = 0; gi < 2; ++gi) {
        const int gt = wave * 2 + gi;
        const bf16_t* brow = P + (size_t)(b * FF + gt * 16 + l16) * FF;
        v8f acc = {};
        for (int k = 0; k < FF; k += 32) {
            Frag a = load_a_frag(arow, k, h);
            Frag bfr = load_b_frag(brow, k, h);
            acc = wmma_bf16(a, bfr, acc);
        }
        const int g = gt * 16 + l16;
        for (int r = 0; r < 8; ++r) {
            const int lt = r + 8 * h;
            const int tok = nt * 16 + lt;
            const size_t idx = (size_t)(b * NN + tok) * FF + g;
            float v = (acc[r] - dvec[b * NN + tok] * unary_f[idx]) * inv_n;
            att[lt * FF + g] = (bf16_t)v;
        }
    }
    __syncthreads();

    // Phase 2: out = att @ Wr^T + x   (A-fragments from LDS -> ds_load_b128)
    const bf16_t* la = att + (size_t)l16 * FF;
    for (int gi = 0; gi < 2; ++gi) {
        const int gt = wave * 2 + gi;
        const bf16_t* brow = Wr + (size_t)(gt * 16 + l16) * FF;
        v8f acc = {};
        for (int k = 0; k < FF; k += 32) {
            Frag a = load_a_frag(la, k, h);
            Frag bfr = load_b_frag(brow, k, h);
            acc = wmma_bf16(a, bfr, acc);
        }
        const int g = gt * 16 + l16;
        for (int r = 0; r < 8; ++r) {
            const int tok = nt * 16 + r + 8 * h;
            const size_t idx = (size_t)(b * NN + tok) * FF + g;
            out[idx] = acc[r] + x[idx];
        }
    }
}

// ---------------------------------------------------------------------------
extern "C" void kernel_launch(void* const* d_in, const int* in_sizes, int n_in,
                              void* d_out, int out_size, void* d_ws,
                              size_t ws_size, hipStream_t stream) {
    const float* x = (const float*)d_in[0];
    const float* Wa = (const float*)d_in[1];
    const float* Wb = (const float*)d_in[2];
    const float* Wu = (const float*)d_in[3];
    const float* bias = (const float*)d_in[4];
    const float* Wr = (const float*)d_in[5];
    float* out = (float*)d_out;

    const size_t XN = (size_t)BB * NN * FF; // 4,194,304
    const size_t WN = (size_t)FF * FF;      // 65,536

    // Bump-allocate workspace (all sizes 256B-aligned by construction).
    char* ws = (char*)d_ws;
    auto alloc = [&](size_t bytes) {
        char* p = ws;
        ws += (bytes + 255) & ~(size_t)255;
        return p;
    };
    bf16_t* xb      = (bf16_t*)alloc(XN * 2);
    bf16_t* Wab     = (bf16_t*)alloc(WN * 2);
    bf16_t* Wbb     = (bf16_t*)alloc(WN * 2);
    bf16_t* Wub     = (bf16_t*)alloc(WN * 2);
    bf16_t* Wrb     = (bf16_t*)alloc(WN * 2);
    bf16_t* alpha   = (bf16_t*)alloc(XN * 2);
    bf16_t* beta    = (bf16_t*)alloc(XN * 2);
    bf16_t* betaT   = (bf16_t*)alloc(XN * 2);
    bf16_t* unaryT  = (bf16_t*)alloc(XN * 2);
    float*  unary_f = (float*)alloc(XN * 4);
    bf16_t* P       = (bf16_t*)alloc((size_t)BB * FF * FF * 2);
    float*  dvec    = (float*)alloc((size_t)BB * NN * 4);

    // K0: convert x + weights to bf16
    cvt_bf16_kernel<<<(int)(XN / 256), 256, 0, stream>>>(x, xb, (int)XN);
    cvt_bf16_kernel<<<(int)(WN / 256), 256, 0, stream>>>(Wa, Wab, (int)WN);
    cvt_bf16_kernel<<<(int)(WN / 256), 256, 0, stream>>>(Wb, Wbb, (int)WN);
    cvt_bf16_kernel<<<(int)(WN / 256), 256, 0, stream>>>(Wu, Wub, (int)WN);
    cvt_bf16_kernel<<<(int)(WN / 256), 256, 0, stream>>>(Wr, Wrb, (int)WN);

    // K1: alpha/beta/unary projections (+ transposed copies)
    proj_kernel<<<dim3(BB * NT, 3), 256, 0, stream>>>(
        xb, Wab, Wbb, Wub, bias, alpha, beta, betaT, unary_f, unaryT);

    // K2: diagonal scores d_i = alpha_i . beta_i
    diag_kernel<<<(BB * NN) / 256, 256, 0, stream>>>(alpha, beta, dvec);

    // K3: P = unary^T @ beta (= M^T) per batch
    pmat_kernel<<<dim3(BB, 32), 256, 0, stream>>>(unaryT, betaT, P);

    // K4: att (diag-corrected, /N) -> LDS -> out = att @ Wr^T + x
    final_kernel<<<BB * NT, 256, 0, stream>>>(alpha, P, dvec, unary_f, Wrb, x,
                                              out);
}
